// GAT_80925773791476
// MI455X (gfx1250) — matrix operand order repaired and checked by reading, not compiled, not run on hardware
//
#include <hip/hip_runtime.h>
#include <hip/hip_bf16.h>
#include <cstdint>

#define N_NODES 50000
#define N_EDGES 400000
#define F_IN    128
#define EMB     64
#define G_GRAPHS 256
#define E2      (N_EDGES + N_NODES)   // edges + self loops = 450000

typedef __attribute__((ext_vector_type(16))) _Float16 v16h;
typedef __attribute__((ext_vector_type(8)))  _Float16 v8h;
typedef __attribute__((ext_vector_type(8)))  float    v8f;

// ---------- helpers ----------

__device__ __forceinline__ void atomicMaxF(float* addr, float val) {
  // monotone-max trick: int compare for >=0, unsigned-min for <0 (init to -inf)
  if (val >= 0.0f) {
    atomicMax((int*)addr, __float_as_int(val));
  } else {
    atomicMin((unsigned int*)addr, __float_as_uint(val));
  }
}

// ---------- kernels ----------

__global__ void gat_build_edges(const int* __restrict__ ei,
                                int* __restrict__ src, int* __restrict__ dst) {
  int i = blockIdx.x * 256 + threadIdx.x;
  if (i >= E2) return;
  if (i < N_EDGES) { src[i] = ei[i]; dst[i] = ei[N_EDGES + i]; }
  else             { src[i] = i - N_EDGES; dst[i] = i - N_EDGES; }
}

__global__ void gat_cvt_f32_f16(const float* __restrict__ x,
                                _Float16* __restrict__ y, int n) {
  int i = blockIdx.x * 256 + threadIdx.x;
  if (i < n) y[i] = (_Float16)x[i];
}

__global__ void gat_fill_f32(float* __restrict__ p, float v, int n) {
  int i = blockIdx.x * 256 + threadIdx.x;
  if (i < n) p[i] = v;
}

// One wave (32 lanes) per 16x16 output tile; dual accumulators (W_l, W_r)
// share the A fragment.  D = X @ W^T with X:[N,K] f16 row-major,
// W:[O,K] f16 row-major.  A/B k order: contiguous, k = 16*hs + e; C/D per CDNA5 ISA.
__global__ void __launch_bounds__(32)
gat_gemm_dual_wmma(const _Float16* __restrict__ Xh,
                   const _Float16* __restrict__ Wl,
                   const _Float16* __restrict__ Wr,
                   float* __restrict__ XL, float* __restrict__ XR,
                   int K, int O) {
  const int lane = threadIdx.x;
  const int m    = lane & 15;     // A row within tile / B-C column within tile
  const int hs   = lane >> 4;     // lane half-select
  const int row  = blockIdx.x * 16 + m;
  const int colT = blockIdx.y * 16;

  v8f accL = {};
  v8f accR = {};
  const _Float16* xrow = Xh + (size_t)row * K;
  const _Float16* wlc  = Wl + (size_t)(colT + m) * K;
  const _Float16* wrc  = Wr + (size_t)(colT + m) * K;

  for (int k0 = 0; k0 < K; k0 += 32) {
    // A fragment: lanes 0-15 hold K = e; lanes 16-31 hold K = 16+e
    v8h alo = *(const v8h*)(xrow + k0 + hs * 16);
    v8h ahi = *(const v8h*)(xrow + k0 + hs * 16 + 8);
    v16h a;
#pragma unroll
    for (int i = 0; i < 8; ++i) { a[i] = alo[i]; a[i + 8] = ahi[i]; }
    // B fragment: lane's 16 halfs are 16 consecutive K values of its column
    v16h bl = *(const v16h*)(wlc + k0 + hs * 16);
    v16h br = *(const v16h*)(wrc + k0 + hs * 16);
    accL = __builtin_amdgcn_wmma_f32_16x16x32_f16(false, a, false, bl,
                                                  (short)0, accL, false, false);
    accR = __builtin_amdgcn_wmma_f32_16x16x32_f16(false, a, false, br,
                                                  (short)0, accR, false, false);
  }
#pragma unroll
  for (int r = 0; r < 8; ++r) {
    int mr = blockIdx.x * 16 + r + hs * 8;   // C/D: VGPR r -> M = r + 8*hs
    XL[(size_t)mr * O + colT + m] = accL[r];
    XR[(size_t)mr * O + colT + m] = accR[r];
  }
}

// Pass 1: per-(edge,head) attention logit + segment max over dst
__global__ void gat_edge_logits_max(const float* __restrict__ XL,
                                    const float* __restrict__ XR,
                                    const float* __restrict__ att,
                                    const int* __restrict__ src,
                                    const int* __restrict__ dst,
                                    float* __restrict__ logits,
                                    float* __restrict__ mbuf,
                                    int O, int H) {
  int i = blockIdx.x * 256 + threadIdx.x;
  if (i >= E2 * H) return;
  int e = i / H, h = i % H;
  int s = src[e], d = dst[e];
  const float4* pl = (const float4*)(XL + (size_t)s * O + h * EMB);
  const float4* pr = (const float4*)(XR + (size_t)d * O + h * EMB);
  const float4* pa = (const float4*)(att + h * EMB);
  float acc = 0.0f;
#pragma unroll 4
  for (int q = 0; q < EMB / 4; ++q) {
    float4 a = pl[q], b = pr[q], w = pa[q];
    float v;
    v = a.x + b.x; v = v > 0.0f ? v : 0.2f * v; acc += v * w.x;
    v = a.y + b.y; v = v > 0.0f ? v : 0.2f * v; acc += v * w.y;
    v = a.z + b.z; v = v > 0.0f ? v : 0.2f * v; acc += v * w.z;
    v = a.w + b.w; v = v > 0.0f ? v : 0.2f * v; acc += v * w.w;
  }
  logits[i] = acc;
  atomicMaxF(&mbuf[d * H + h], acc);
}

// Pass 2: p = exp(e - m[dst]); segment sum over dst
__global__ void gat_edge_softmax_num(float* __restrict__ logits,
                                     const float* __restrict__ mbuf,
                                     float* __restrict__ ssum,
                                     const int* __restrict__ dst, int H) {
  int i = blockIdx.x * 256 + threadIdx.x;
  if (i >= E2 * H) return;
  int e = i / H, h = i % H;
  int d = dst[e];
  float p = __expf(logits[i] - mbuf[d * H + h]);
  logits[i] = p;
  atomicAdd(&ssum[d * H + h], p);
}

// Pass 3: agg[dst] += alpha * xl[src]  (per edge, per output feature)
__global__ void gat_edge_scatter(const float* __restrict__ logits,
                                 const float* __restrict__ ssum,
                                 const float* __restrict__ XL,
                                 const int* __restrict__ src,
                                 const int* __restrict__ dst,
                                 float* __restrict__ agg,
                                 int O, int H) {
  long long i = (long long)blockIdx.x * 256 + threadIdx.x;
  if (i >= (long long)E2 * O) return;
  int e = (int)(i / O);
  int o = (int)(i % O);
  int h = o / EMB;
  int s = src[e], d = dst[e];
  float alpha = logits[(size_t)e * H + h] / (ssum[d * H + h] + 1e-16f);
  atomicAdd(&agg[(size_t)d * O + o], alpha * XL[(size_t)s * O + o]);
}

// y = leaky_relu(y + bias, 0.01) in place
__global__ void gat_bias_act(float* __restrict__ y,
                             const float* __restrict__ bias, int O) {
  int i = blockIdx.x * 256 + threadIdx.x;
  if (i >= N_NODES * O) return;
  float v = y[i] + bias[i % O];
  y[i] = v > 0.0f ? v : 0.01f * v;
}

// per-graph max + sum pooling (h: [N, EMB])
__global__ void gat_pool(const float* __restrict__ h,
                         const int* __restrict__ batch,
                         float* __restrict__ gmax,
                         float* __restrict__ gsum,
                         float* __restrict__ cnt) {
  int i = blockIdx.x * 256 + threadIdx.x;
  if (i >= N_NODES * EMB) return;
  int node = i / EMB, d = i % EMB;
  int g = batch[node];
  float v = h[i];
  atomicMaxF(&gmax[g * EMB + d], v);
  atomicAdd(&gsum[g * EMB + d], v);
  if (d == 0) atomicAdd(&cnt[g], 1.0f);
}

// hidden = [gmax, gmean]; out = hidden @ out_w^T + out_b
__global__ void gat_head(const float* __restrict__ gmax,
                         const float* __restrict__ gsum,
                         const float* __restrict__ cnt,
                         const float* __restrict__ out_w,
                         const float* __restrict__ out_b,
                         float* __restrict__ out) {
  int g = blockIdx.x * 256 + threadIdx.x;
  if (g >= G_GRAPHS) return;
  float inv = 1.0f / fmaxf(cnt[g], 1.0f);
  float hid[2 * EMB];
#pragma unroll
  for (int d = 0; d < EMB; ++d) {
    hid[d]       = gmax[g * EMB + d];
    hid[EMB + d] = gsum[g * EMB + d] * inv;
  }
  float* hidden = out + G_GRAPHS * 2;           // tuple part 2
#pragma unroll
  for (int d = 0; d < 2 * EMB; ++d) hidden[g * 2 * EMB + d] = hid[d];
  for (int j = 0; j < 2; ++j) {
    float acc = out_b[j];
    for (int d = 0; d < 2 * EMB; ++d) acc += hid[d] * out_w[j * 2 * EMB + d];
    out[g * 2 + j] = acc;
  }
}

// ---------- driver ----------

extern "C" void kernel_launch(void* const* d_in, const int* in_sizes, int n_in,
                              void* d_out, int out_size, void* d_ws, size_t ws_size,
                              hipStream_t stream) {
  const float* x      = (const float*)d_in[0];
  const int*   ei     = (const int*)d_in[1];
  const int*   batch  = (const int*)d_in[2];
  const float* w1l = (const float*)d_in[3],  *w1r = (const float*)d_in[4];
  const float* att1 = (const float*)d_in[5], *b1 = (const float*)d_in[6];
  const float* w2l = (const float*)d_in[7],  *w2r = (const float*)d_in[8];
  const float* att2 = (const float*)d_in[9], *b2 = (const float*)d_in[10];
  const float* w3l = (const float*)d_in[11], *w3r = (const float*)d_in[12];
  const float* att3 = (const float*)d_in[13],*b3 = (const float*)d_in[14];
  const float* out_w = (const float*)d_in[15], *out_b = (const float*)d_in[16];
  float* out = (float*)d_out;

  // workspace carve (256B aligned)
  char* wp = (char*)d_ws;
  auto take = [&](size_t bytes) -> void* {
    void* r = (void*)wp;
    wp += (bytes + 255) & ~(size_t)255;
    return r;
  };
  int*      srcA  = (int*)take((size_t)E2 * 4);
  int*      dstA  = (int*)take((size_t)E2 * 4);
  _Float16* Xh    = (_Float16*)take((size_t)N_NODES * 256 * 2);
  _Float16* Wlh   = (_Float16*)take((size_t)256 * 256 * 2);
  _Float16* Wrh   = (_Float16*)take((size_t)256 * 256 * 2);
  float*    XL    = (float*)take((size_t)N_NODES * 256 * 4);
  float*    XR    = (float*)take((size_t)N_NODES * 256 * 4);
  float*    logit = (float*)take((size_t)E2 * 4 * 4);
  float*    mbuf  = (float*)take((size_t)N_NODES * 4 * 4);
  float*    ssum  = (float*)take((size_t)N_NODES * 4 * 4);
  float*    agg1  = (float*)take((size_t)N_NODES * 256 * 4);
  float*    agg2  = (float*)take((size_t)N_NODES * 128 * 4);
  float*    agg3  = (float*)take((size_t)N_NODES * 64 * 4);
  float*    gmax  = (float*)take((size_t)G_GRAPHS * EMB * 4);
  float*    gsum  = (float*)take((size_t)G_GRAPHS * EMB * 4);
  float*    cnt   = (float*)take((size_t)G_GRAPHS * 4);

  const float NEG_INF = -__builtin_huge_valf();

  gat_build_edges<<<(E2 + 255) / 256, 256, 0, stream>>>(ei, srcA, dstA);

  auto run_layer = [&](const float* hin, int K, int O, int H,
                       const float* wl, const float* wr,
                       const float* att, const float* bias, float* agg) {
    int nk = N_NODES * K;
    gat_cvt_f32_f16<<<(nk + 255) / 256, 256, 0, stream>>>(hin, Xh, nk);
    gat_cvt_f32_f16<<<(O * K + 255) / 256, 256, 0, stream>>>(wl, Wlh, O * K);
    gat_cvt_f32_f16<<<(O * K + 255) / 256, 256, 0, stream>>>(wr, Wrh, O * K);

    dim3 gg(N_NODES / 16, O / 16);
    gat_gemm_dual_wmma<<<gg, 32, 0, stream>>>(Xh, Wlh, Wrh, XL, XR, K, O);

    gat_fill_f32<<<(N_NODES * H + 255) / 256, 256, 0, stream>>>(mbuf, NEG_INF,
                                                                N_NODES * H);
    hipMemsetAsync(ssum, 0, (size_t)N_NODES * H * 4, stream);
    hipMemsetAsync(agg, 0, (size_t)N_NODES * O * 4, stream);

    int eh = E2 * H;
    gat_edge_logits_max<<<(eh + 255) / 256, 256, 0, stream>>>(
        XL, XR, att, srcA, dstA, logit, mbuf, O, H);
    gat_edge_softmax_num<<<(eh + 255) / 256, 256, 0, stream>>>(
        logit, mbuf, ssum, dstA, H);
    long long eo = (long long)E2 * O;
    gat_edge_scatter<<<(unsigned)((eo + 255) / 256), 256, 0, stream>>>(
        logit, ssum, XL, srcA, dstA, agg, O, H);
    gat_bias_act<<<(N_NODES * O + 255) / 256, 256, 0, stream>>>(agg, bias, O);
  };

  run_layer(x,    F_IN,     4 * EMB, 4, w1l, w1r, att1, b1, agg1);
  run_layer(agg1, 4 * EMB,  2 * EMB, 2, w2l, w2r, att2, b2, agg2);
  run_layer(agg2, 2 * EMB,  1 * EMB, 1, w3l, w3r, att3, b3, agg3);

  // pooling + head
  gat_fill_f32<<<(G_GRAPHS * EMB + 255) / 256, 256, 0, stream>>>(
      gmax, NEG_INF, G_GRAPHS * EMB);
  hipMemsetAsync(gsum, 0, (size_t)G_GRAPHS * EMB * 4, stream);
  hipMemsetAsync(cnt, 0, (size_t)G_GRAPHS * 4, stream);
  gat_pool<<<(N_NODES * EMB + 255) / 256, 256, 0, stream>>>(agg3, batch, gmax,
                                                            gsum, cnt);
  gat_head<<<1, 256, 0, stream>>>(gmax, gsum, cnt, out_w, out_b, out);
}
